// sLSTMCell_28226525070218
// MI455X (gfx1250) — matrix operand order repaired and checked by reading, not compiled, not run on hardware
//
#include <hip/hip_runtime.h>
#include <hip/hip_bf16.h>
#include <math.h>
#include <stdint.h>

#define B_DIM 16384
#define D_IN  512
#define H_DIM 1024

#define LDA   40            // bf16 elements per padded LDS row (80 bytes)
#define ASEG  (64 * LDA)    // one 64x32 tile = 2560 bf16
#define BSEG  (64 * LDA)    // one gate's 64x32 tile

typedef __attribute__((ext_vector_type(16))) __bf16 v16bf;
typedef __attribute__((ext_vector_type(8)))  __bf16 v8bf;
typedef __attribute__((ext_vector_type(8)))  float  v8f;

__device__ __forceinline__ __bf16 f2bf(float f) { return (__bf16)f; }

__device__ __forceinline__ v16bf cat8(v8bf lo, v8bf up) {
  return __builtin_shufflevector(lo, up, 0, 1, 2, 3, 4, 5, 6, 7,
                                 8, 9, 10, 11, 12, 13, 14, 15);
}

// ---------------------------------------------------------------------------
// Pre-pass: fp32 -> bf16, 8 elements per thread (all array sizes % 8 == 0).
// ---------------------------------------------------------------------------
__global__ __launch_bounds__(256)
void cvt_bf16_kernel(const float* __restrict__ src, __bf16* __restrict__ dst, int n) {
  int i = (blockIdx.x * 256 + threadIdx.x) * 8;
  if (i >= n) return;
  const float4* q = (const float4*)(src + i);
  float4 a = q[0], b = q[1];
  v8bf v;
  v[0] = f2bf(a.x); v[1] = f2bf(a.y); v[2] = f2bf(a.z); v[3] = f2bf(a.w);
  v[4] = f2bf(b.x); v[5] = f2bf(b.y); v[6] = f2bf(b.z); v[7] = f2bf(b.w);
  *(v8bf*)(dst + i) = v;
}

// ---------------------------------------------------------------------------
// Fast path: async-to-LDS double-buffered bf16 WMMA GEMM + fused sLSTM tail.
// Block = 256 threads = 8 waves (4 along M x 2 along N); block tile 64x64.
// Per K-step (K=32): A tile 64x32, B tiles 4x(64x32) staged via
// global_load_async_to_lds_b128; compute loads ALL fragments first, then
// issues 8 back-to-back WMMAs (avoids per-WMMA s_wait_dscnt serialization).
// ---------------------------------------------------------------------------
__global__ __launch_bounds__(256)
void slstm_wmma_async(const __bf16* __restrict__ xb,
                      const __bf16* __restrict__ hb,
                      const __bf16* __restrict__ Wb,   // [4][H][DIN]
                      const __bf16* __restrict__ Rb,   // [4][H][H]
                      const float* __restrict__ c_prev,
                      const float* __restrict__ n_prev,
                      const float* __restrict__ bz, const float* __restrict__ bi,
                      const float* __restrict__ bf_, const float* __restrict__ bo,
                      const float* __restrict__ xbz, const float* __restrict__ xbi,
                      const float* __restrict__ xbf, const float* __restrict__ xbo,
                      float* __restrict__ out) {
  __shared__ __align__(16) __bf16 smA[2][ASEG];
  __shared__ __align__(16) __bf16 smB[2][4 * BSEG];

  const int tid    = threadIdx.x;
  const int lane   = tid & 31;
  const int wave   = tid >> 5;
  const int laneLo = lane & 15;
  const int hi     = lane >> 4;
  const int waveM  = wave & 3;
  const int waveN  = wave >> 2;

  const int blockRow0 = blockIdx.x * 64;
  const int blockCol0 = blockIdx.y * 64;
  const int m0 = blockRow0 + waveM * 16;
  const int n0 = blockCol0 + waveN * 32;

  // --- staging constants: thread -> (row 0..63, 16B chunk 0..3) ---
  const int sr = tid >> 2;
  const int sc = tid & 3;
  const unsigned ldsA0 = (unsigned)(uintptr_t)&smA[0][0];
  const unsigned ldsA1 = (unsigned)(uintptr_t)&smA[1][0];
  const unsigned ldsB0 = (unsigned)(uintptr_t)&smB[0][0];
  const unsigned ldsB1 = (unsigned)(uintptr_t)&smB[1][0];
  const unsigned stOff = (unsigned)(sr * LDA + sc * 8) * 2;  // bytes in tile

  v8f acc[8];
#pragma unroll
  for (int t = 0; t < 8; ++t)
    acc[t] = (v8f){0.f, 0.f, 0.f, 0.f, 0.f, 0.f, 0.f, 0.f};

  // Issue one K-step of async global->LDS copies into buffer `buf`.
  auto stage = [&](int buf, const __bf16* gA, const __bf16* gB, int ld, int kk) {
    const unsigned la = (buf ? ldsA1 : ldsA0) + stOff;
    const __bf16* pa = gA + (size_t)(blockRow0 + sr) * ld + kk + sc * 8;
    asm volatile("global_load_async_to_lds_b128 %0, %1, off"
                 :: "v"(la), "v"((unsigned long long)(uintptr_t)pa) : "memory");
#pragma unroll
    for (int g = 0; g < 4; ++g) {
      const unsigned lb = (buf ? ldsB1 : ldsB0) + (unsigned)g * (BSEG * 2) + stOff;
      const __bf16* pb = gB + ((size_t)g * H_DIM + (blockCol0 + sr)) * ld + kk + sc * 8;
      asm volatile("global_load_async_to_lds_b128 %0, %1, off"
                   :: "v"(lb), "v"((unsigned long long)(uintptr_t)pb) : "memory");
    }
  };

  // One K-step: load A + all 8 B fragments from LDS, then 8 WMMAs.
  auto compute = [&](int buf) {
    const int ra = (waveM * 16 + laneLo) * LDA;
    v16bf a = cat8(*(const v8bf*)&smA[buf][ra + hi * 16],
                   *(const v8bf*)&smA[buf][ra + hi * 16 + 8]);
    v16bf bfr[8];
#pragma unroll
    for (int g = 0; g < 4; ++g) {
#pragma unroll
      for (int nc = 0; nc < 2; ++nc) {
        const int cb = g * BSEG + (waveN * 32 + nc * 16 + laneLo) * LDA + hi * 16;
        bfr[g * 2 + nc] = cat8(*(const v8bf*)&smB[buf][cb],
                               *(const v8bf*)&smB[buf][cb + 8]);
      }
    }
#pragma unroll
    for (int t = 0; t < 8; ++t)
      acc[t] = __builtin_amdgcn_wmma_f32_16x16x32_bf16(
          false, a, false, bfr[t], (short)0, acc[t], false, false);
  };

  auto sync_step = [&]() {
    asm volatile("s_wait_asynccnt 0x0" ::: "memory");
    __syncthreads();
  };

  // ---- software pipeline: 16 x/W steps + 32 h/R steps, double buffered ----
  stage(0, xb, Wb, D_IN, 0);
  sync_step();

  for (int step = 0; step < 15; ++step) {          // next tile from x/W
    stage((step + 1) & 1, xb, Wb, D_IN, (step + 1) * 32);
    compute(step & 1);
    sync_step();
  }
  // boundary: step 15 computes, next tile is first h/R tile (buffer 0)
  stage(0, hb, Rb, H_DIM, 0);
  compute(1);
  sync_step();

  for (int step = 16; step < 47; ++step) {         // next tile from h/R
    stage((step + 1) & 1, hb, Rb, H_DIM, (step + 1 - 16) * 32);
    compute(step & 1);
    sync_step();
  }
  compute(1);                                      // step 47, drain

  // ---- elementwise tail (C/D layout: VGPR r -> M=r / M=8+r; lane%16 -> N) ----
  const size_t BH = (size_t)B_DIM * H_DIM;
#pragma unroll
  for (int nc = 0; nc < 2; ++nc) {
    const int h = n0 + nc * 16 + laneLo;
    const float bzv = bz[h]  + xbz[h];
    const float biv = bi[h]  + xbi[h];
    const float bfv = bf_[h] + xbf[h];
    const float bov = bo[h]  + xbo[h];
#pragma unroll
    for (int r = 0; r < 8; ++r) {
      const int m = m0 + 8 * hi + r;
      const size_t idx = (size_t)m * H_DIM + h;
      const float pz = acc[0 * 2 + nc][r] + bzv;
      const float pi = acc[1 * 2 + nc][r] + biv;
      const float pf = acc[2 * 2 + nc][r] + bfv;
      const float po = acc[3 * 2 + nc][r] + bov;
      const float zv = tanhf(pz);
      const float iv = __expf(pi);
      const float fv = 1.0f / (1.0f + __expf(-pf));
      const float ov = 1.0f / (1.0f + __expf(-po));
      const float cp = c_prev[idx];
      const float np = n_prev[idx];
      const float ct = fv * cp + iv * zv;
      const float nt = fv * np + iv;
      const float ht = ov * (ct / nt);
      out[idx]          = ct;
      out[BH + idx]     = nt;
      out[2 * BH + idx] = ht;
    }
  }
}

// ---------------------------------------------------------------------------
// Fallback path (no workspace): fp32 loads + in-loop cvt, direct from global.
// ---------------------------------------------------------------------------
__device__ __forceinline__ v16bf load_frag_A_f32(const float* __restrict__ p, int hi) {
  const float4* q = (const float4*)p;
  float4 a0 = q[hi * 4 + 0], a1 = q[hi * 4 + 1];
  float4 a2 = q[hi * 4 + 2], a3 = q[hi * 4 + 3];
  v16bf v;
  v[0]  = f2bf(a0.x); v[1]  = f2bf(a0.y); v[2]  = f2bf(a0.z); v[3]  = f2bf(a0.w);
  v[4]  = f2bf(a1.x); v[5]  = f2bf(a1.y); v[6]  = f2bf(a1.z); v[7]  = f2bf(a1.w);
  v[8]  = f2bf(a2.x); v[9]  = f2bf(a2.y); v[10] = f2bf(a2.z); v[11] = f2bf(a2.w);
  v[12] = f2bf(a3.x); v[13] = f2bf(a3.y); v[14] = f2bf(a3.z); v[15] = f2bf(a3.w);
  return v;
}
__device__ __forceinline__ v16bf load_frag_B_f32(const float* __restrict__ p) {
  const float4* q = (const float4*)p;
  float4 a0 = q[0], a1 = q[1], a2 = q[2], a3 = q[3];
  v16bf v;
  v[0]  = f2bf(a0.x); v[1]  = f2bf(a0.y); v[2]  = f2bf(a0.z); v[3]  = f2bf(a0.w);
  v[4]  = f2bf(a1.x); v[5]  = f2bf(a1.y); v[6]  = f2bf(a1.z); v[7]  = f2bf(a1.w);
  v[8]  = f2bf(a2.x); v[9]  = f2bf(a2.y); v[10] = f2bf(a2.z); v[11] = f2bf(a2.w);
  v[12] = f2bf(a3.x); v[13] = f2bf(a3.y); v[14] = f2bf(a3.z); v[15] = f2bf(a3.w);
  return v;
}

__global__ __launch_bounds__(256)
void slstm_fused_f32(const float* __restrict__ x, const float* __restrict__ c_prev,
                     const float* __restrict__ n_prev, const float* __restrict__ h_prev,
                     const float* __restrict__ Wz, const float* __restrict__ Wi,
                     const float* __restrict__ Wf, const float* __restrict__ Wo,
                     const float* __restrict__ bz, const float* __restrict__ bi,
                     const float* __restrict__ bf_, const float* __restrict__ bo,
                     const float* __restrict__ Rz, const float* __restrict__ Ri,
                     const float* __restrict__ Rf, const float* __restrict__ Ro,
                     const float* __restrict__ xbz, const float* __restrict__ xbi,
                     const float* __restrict__ xbf, const float* __restrict__ xbo,
                     float* __restrict__ out) {
  const int lane   = threadIdx.x & 31;
  const int wave   = threadIdx.x >> 5;
  const int laneLo = lane & 15;
  const int hi     = lane >> 4;
  const int waveM  = wave & 3;
  const int waveN  = wave >> 2;
  const int m0 = blockIdx.x * 64 + waveM * 16;
  const int n0 = blockIdx.y * 64 + waveN * 32;
  const int rowA = m0 + laneLo;
  const int hB0  = n0 + laneLo;
  const int hB1  = n0 + 16 + laneLo;

  v8f acc[8];
#pragma unroll
  for (int t = 0; t < 8; ++t)
    acc[t] = (v8f){0.f, 0.f, 0.f, 0.f, 0.f, 0.f, 0.f, 0.f};

  const float* Wg[4] = {Wz, Wi, Wf, Wo};
  const float* Rg[4] = {Rz, Ri, Rf, Ro};

  {
    const float* arow = x + (size_t)rowA * D_IN;
    for (int kk = 0; kk < D_IN; kk += 32) {
      v16bf a = load_frag_A_f32(arow + kk, hi);
      const int kb = kk + hi * 16;
#pragma unroll
      for (int g = 0; g < 4; ++g) {
        v16bf b0 = load_frag_B_f32(Wg[g] + (size_t)hB0 * D_IN + kb);
        acc[g * 2 + 0] = __builtin_amdgcn_wmma_f32_16x16x32_bf16(
            false, a, false, b0, (short)0, acc[g * 2 + 0], false, false);
        v16bf b1 = load_frag_B_f32(Wg[g] + (size_t)hB1 * D_IN + kb);
        acc[g * 2 + 1] = __builtin_amdgcn_wmma_f32_16x16x32_bf16(
            false, a, false, b1, (short)0, acc[g * 2 + 1], false, false);
      }
    }
  }
  {
    const float* arow = h_prev + (size_t)rowA * H_DIM;
    for (int kk = 0; kk < H_DIM; kk += 32) {
      v16bf a = load_frag_A_f32(arow + kk, hi);
      const int kb = kk + hi * 16;
#pragma unroll
      for (int g = 0; g < 4; ++g) {
        v16bf b0 = load_frag_B_f32(Rg[g] + (size_t)hB0 * H_DIM + kb);
        acc[g * 2 + 0] = __builtin_amdgcn_wmma_f32_16x16x32_bf16(
            false, a, false, b0, (short)0, acc[g * 2 + 0], false, false);
        v16bf b1 = load_frag_B_f32(Rg[g] + (size_t)hB1 * H_DIM + kb);
        acc[g * 2 + 1] = __builtin_amdgcn_wmma_f32_16x16x32_bf16(
            false, a, false, b1, (short)0, acc[g * 2 + 1], false, false);
      }
    }
  }

  const size_t BH = (size_t)B_DIM * H_DIM;
#pragma unroll
  for (int nc = 0; nc < 2; ++nc) {
    const int h = n0 + nc * 16 + laneLo;
    const float bzv = bz[h]  + xbz[h];
    const float biv = bi[h]  + xbi[h];
    const float bfv = bf_[h] + xbf[h];
    const float bov = bo[h]  + xbo[h];
#pragma unroll
    for (int r = 0; r < 8; ++r) {
      const int m = m0 + 8 * hi + r;
      const size_t idx = (size_t)m * H_DIM + h;
      const float pz = acc[0 * 2 + nc][r] + bzv;
      const float pi = acc[1 * 2 + nc][r] + biv;
      const float pf = acc[2 * 2 + nc][r] + bfv;
      const float po = acc[3 * 2 + nc][r] + bov;
      const float zv = tanhf(pz);
      const float iv = __expf(pi);
      const float fv = 1.0f / (1.0f + __expf(-pf));
      const float ov = 1.0f / (1.0f + __expf(-po));
      const float cp = c_prev[idx];
      const float np = n_prev[idx];
      const float ct = fv * cp + iv * zv;
      const float nt = fv * np + iv;
      const float ht = ov * (ct / nt);
      out[idx]          = ct;
      out[BH + idx]     = nt;
      out[2 * BH + idx] = ht;
    }
  }
}

// ---------------------------------------------------------------------------
extern "C" void kernel_launch(void* const* d_in, const int* in_sizes, int n_in,
                              void* d_out, int out_size, void* d_ws, size_t ws_size,
                              hipStream_t stream) {
  const float* x      = (const float*)d_in[0];
  const float* c_prev = (const float*)d_in[1];
  const float* n_prev = (const float*)d_in[2];
  const float* h_prev = (const float*)d_in[3];
  const float* W[4]   = {(const float*)d_in[4], (const float*)d_in[5],
                         (const float*)d_in[6], (const float*)d_in[7]};
  const float* bz     = (const float*)d_in[8];
  const float* bi     = (const float*)d_in[9];
  const float* bf_    = (const float*)d_in[10];
  const float* bo     = (const float*)d_in[11];
  const float* R[4]   = {(const float*)d_in[12], (const float*)d_in[13],
                         (const float*)d_in[14], (const float*)d_in[15]};
  const float* xbz    = (const float*)d_in[16];
  const float* xbi    = (const float*)d_in[17];
  const float* xbf    = (const float*)d_in[18];
  const float* xbo    = (const float*)d_in[19];
  float* out = (float*)d_out;

  const size_t nX = (size_t)B_DIM * D_IN;
  const size_t nH = (size_t)B_DIM * H_DIM;
  const size_t nW = (size_t)H_DIM * D_IN;
  const size_t nR = (size_t)H_DIM * H_DIM;
  const size_t needBytes = (nX + nH + 4 * nW + 4 * nR) * sizeof(__bf16); // ~60MB

  dim3 grid(B_DIM / 64, H_DIM / 64);
  dim3 block(256);

  if (ws_size >= needBytes) {
    __bf16* xb = (__bf16*)d_ws;
    __bf16* hb = xb + nX;
    __bf16* Wb = hb + nH;
    __bf16* Rb = Wb + 4 * nW;

    auto cvt = [&](const float* src, __bf16* dst, size_t n) {
      int blocks = (int)((n / 8 + 255) / 256);
      cvt_bf16_kernel<<<blocks, 256, 0, stream>>>(src, dst, (int)n);
    };
    cvt(x, xb, nX);
    cvt(h_prev, hb, nH);
    for (int g = 0; g < 4; ++g) cvt(W[g], Wb + g * nW, nW);
    for (int g = 0; g < 4; ++g) cvt(R[g], Rb + g * nR, nR);

    slstm_wmma_async<<<grid, block, 0, stream>>>(
        xb, hb, Wb, Rb, c_prev, n_prev,
        bz, bi, bf_, bo, xbz, xbi, xbf, xbo, out);
  } else {
    slstm_fused_f32<<<grid, block, 0, stream>>>(
        x, c_prev, n_prev, h_prev,
        W[0], W[1], W[2], W[3], bz, bi, bf_, bo,
        R[0], R[1], R[2], R[3], xbz, xbi, xbf, xbo, out);
  }
}
